// GATConv_67748814127735
// MI455X (gfx1250) — compile-verified, matrix operand order checked by reading, not run on hardware
//
#include <hip/hip_runtime.h>
#include <stdint.h>

#define N_NODES   100000
#define N_EDGES   1600000
#define IN_FEATS  128
#define NUM_HEADS 4
#define OUT_FEATS 32
#define HD        128            // NUM_HEADS * OUT_FEATS
#define NEG_SLOPE 0.2f

typedef __attribute__((ext_vector_type(16))) __bf16 bf16x16;
typedef __attribute__((ext_vector_type(8)))  float  f32x8;

// ---------- helpers ----------
__device__ __forceinline__ unsigned short f32_to_bf16(float f) {
  unsigned int u = __float_as_uint(f);
  u = (u + 0x7FFFu + ((u >> 16) & 1u)) >> 16;   // RNE
  return (unsigned short)u;
}
// order-preserving float -> uint key for atomicMax-based segment max
__device__ __forceinline__ unsigned int f2key(float f) {
  unsigned int b = __float_as_uint(f);
  return (b & 0x80000000u) ? ~b : (b | 0x80000000u);
}
__device__ __forceinline__ float key2f(unsigned int k) {
  return (k & 0x80000000u) ? __uint_as_float(k ^ 0x80000000u)
                           : __uint_as_float(~k);
}

// ---------- 1) ft = feat @ W  via v_wmma_f32_16x16x32_bf16 ----------
// Block: 128 threads = 4 waves; each wave owns a 16-row stripe -> 64 rows/block.
// LDS: A tile (64x128 bf16) + W^T (128x128 bf16), stride padded to 136 halves.
#define LDA 136
__global__ __launch_bounds__(128) void gat_gemm_wmma(
    const float* __restrict__ feat, const float* __restrict__ W,
    float* __restrict__ ft) {
  __shared__ unsigned short sA[64 * LDA];
  __shared__ unsigned short sWt[128 * LDA];
  const int tid  = threadIdx.x;
  const int row0 = blockIdx.x * 64;

  // stage W^T as bf16: sWt[n][k] = W[k][n]  (coalesced global reads)
  for (int idx = tid; idx < 128 * 128; idx += 128) {
    int k = idx >> 7, n = idx & 127;
    sWt[n * LDA + k] = f32_to_bf16(W[idx]);
  }
  // stage A rows as bf16
  for (int idx = tid; idx < 64 * 128; idx += 128) {
    int r = idx >> 7, c = idx & 127;
    int row = row0 + r;
    float v = (row < N_NODES) ? feat[(size_t)row * IN_FEATS + c] : 0.0f;
    sA[r * LDA + c] = f32_to_bf16(v);
  }
  __syncthreads();

  const int wid  = tid >> 5;       // wave id 0..3
  const int lane = tid & 31;
  const int m    = lane & 15;
  const int half = lane >> 4;      // ISA 16-bit frag layout: half selects K groups

  f32x8 acc[8];
#pragma unroll
  for (int nt = 0; nt < 8; ++nt) acc[nt] = (f32x8)(0.0f);

  const int rowA = wid * 16 + m;
#pragma unroll
  for (int kb = 0; kb < 128; kb += 32) {
    // A fragment: lane (half,m): K = kb + {half*8..half*8+7, 16+half*8..}
    bf16x16 a;
    ((uint4*)&a)[0] = *(const uint4*)&sA[rowA * LDA + kb + half * 8];
    ((uint4*)&a)[1] = *(const uint4*)&sA[rowA * LDA + kb + 16 + half * 8];
#pragma unroll
    for (int nt = 0; nt < 8; ++nt) {
      const int ncol = nt * 16 + m;      // B fragment from W^T (K contiguous)
      bf16x16 b;
      ((uint4*)&b)[0] = *(const uint4*)&sWt[ncol * LDA + kb + half * 8];
      ((uint4*)&b)[1] = *(const uint4*)&sWt[ncol * LDA + kb + 16 + half * 8];
      acc[nt] = __builtin_amdgcn_wmma_f32_16x16x32_bf16(
          false, a, false, b, (short)0, acc[nt], false, false);
    }
  }

  // C/D layout: VGPR r -> M = r + half*8 (within wave's 16-row stripe), N = nt*16 + m
#pragma unroll
  for (int nt = 0; nt < 8; ++nt) {
    const int ncol = nt * 16 + m;
#pragma unroll
    for (int r = 0; r < 8; ++r) {
      int row = row0 + wid * 16 + half * 8 + r;
      if (row < N_NODES) ft[(size_t)row * HD + ncol] = acc[nt][r];
    }
  }
}

// ---------- 2) el/er per (node, head) ----------
__global__ __launch_bounds__(256) void gat_elr(
    const float* __restrict__ ft, const float* __restrict__ attn_l,
    const float* __restrict__ attn_r, float* __restrict__ el,
    float* __restrict__ er) {
  __shared__ float sl[HD], sr[HD];
  if (threadIdx.x < HD) {
    sl[threadIdx.x] = attn_l[threadIdx.x];
    sr[threadIdx.x] = attn_r[threadIdx.x];
  }
  __syncthreads();
  int gid = blockIdx.x * blockDim.x + threadIdx.x;
  if (gid >= N_NODES * NUM_HEADS) return;
  int n = gid >> 2, h = gid & 3;
  const float* f  = ft + (size_t)n * HD + h * OUT_FEATS;
  const float* al = sl + h * OUT_FEATS;
  const float* ar = sr + h * OUT_FEATS;
  float sL = 0.f, sR = 0.f;
#pragma unroll
  for (int d = 0; d < OUT_FEATS; ++d) {
    float v = f[d];
    sL += v * al[d];
    sR += v * ar[d];
  }
  el[gid] = sL;
  er[gid] = sR;
}

// ---------- 3) init segment-max keys / segment sums ----------
__global__ void gat_init(unsigned int* __restrict__ emaxk,
                         float* __restrict__ esum) {
  int gid = blockIdx.x * blockDim.x + threadIdx.x;
  if (gid < N_NODES * NUM_HEADS) { emaxk[gid] = 0u; esum[gid] = 0.0f; }
}

// ---------- 4) edge scores + segment max (atomic u32 max on monotone key) ----------
__global__ __launch_bounds__(256) void gat_edge_score(
    const int* __restrict__ src, const int* __restrict__ dst,
    const float* __restrict__ el, const float* __restrict__ er,
    float* __restrict__ esc, unsigned int* __restrict__ emaxk) {
  int e = blockIdx.x * blockDim.x + threadIdx.x;
  if (e >= N_EDGES) return;
  int s = src[e], d = dst[e];
#pragma unroll
  for (int h = 0; h < NUM_HEADS; ++h) {
    float v = el[s * NUM_HEADS + h] + er[d * NUM_HEADS + h];
    v = v > 0.0f ? v : NEG_SLOPE * v;
    esc[(size_t)e * NUM_HEADS + h] = v;
    atomicMax(&emaxk[d * NUM_HEADS + h], f2key(v));
  }
}

// ---------- 5) ex = exp(e - max), segment sum (global_atomic_add_f32) ----------
__global__ __launch_bounds__(256) void gat_edge_exp(
    const int* __restrict__ dst, const unsigned int* __restrict__ emaxk,
    float* __restrict__ esc, float* __restrict__ esum) {
  int e = blockIdx.x * blockDim.x + threadIdx.x;
  if (e >= N_EDGES) return;
  int d = dst[e];
#pragma unroll
  for (int h = 0; h < NUM_HEADS; ++h) {
    float mx = key2f(emaxk[d * NUM_HEADS + h]);
    float ex = __expf(esc[(size_t)e * NUM_HEADS + h] - mx);
    esc[(size_t)e * NUM_HEADS + h] = ex;               // in-place: scores -> exp
    atomicAdd(&esum[d * NUM_HEADS + h], ex);
  }
}

// ---------- 6) aggregation: one wave32 per destination node, no atomics ----------
// Lane owns components (h, d=lane) for h=0..3; dst is sorted so each node's
// edges are a contiguous range found by binary search.
__global__ __launch_bounds__(256) void gat_aggregate(
    const int* __restrict__ src, const int* __restrict__ dst,
    const float* __restrict__ ft, const float* __restrict__ esc,
    const float* __restrict__ esum, float* __restrict__ out) {
  int node = (blockIdx.x * blockDim.x + threadIdx.x) >> 5;
  int lane = threadIdx.x & 31;
  if (node >= N_NODES) return;

  // wave-uniform binary searches for [beg, end)
  int lo = 0, hi = N_EDGES;
  while (lo < hi) { int mid = (lo + hi) >> 1; if (dst[mid] < node) lo = mid + 1; else hi = mid; }
  const int beg = lo;
  hi = N_EDGES;
  while (lo < hi) { int mid = (lo + hi) >> 1; if (dst[mid] <= node) lo = mid + 1; else hi = mid; }
  const int end = lo;

  float inv[NUM_HEADS];
#pragma unroll
  for (int h = 0; h < NUM_HEADS; ++h) {
    float s = esum[node * NUM_HEADS + h];
    inv[h] = (s > 0.0f) ? 1.0f / s : 0.0f;
  }

  float acc[NUM_HEADS] = {0.f, 0.f, 0.f, 0.f};
  for (int e = beg; e < end; ++e) {
    int s = src[e];
    if (e + 1 < end)  // cover gather latency of the next ft row
      __builtin_prefetch(ft + (size_t)src[e + 1] * HD + lane, 0, 0);
    const float* fs = ft + (size_t)s * HD;
#pragma unroll
    for (int h = 0; h < NUM_HEADS; ++h) {
      float w = esc[(size_t)e * NUM_HEADS + h] * inv[h];
      acc[h] += w * fs[h * OUT_FEATS + lane];   // coalesced 128B per h
    }
  }
#pragma unroll
  for (int h = 0; h < NUM_HEADS; ++h)
    out[(size_t)node * HD + h * OUT_FEATS + lane] = acc[h];
}

// ---------- launch ----------
extern "C" void kernel_launch(void* const* d_in, const int* in_sizes, int n_in,
                              void* d_out, int out_size, void* d_ws, size_t ws_size,
                              hipStream_t stream) {
  const float* feat   = (const float*)d_in[0];
  const int*   src    = (const int*)  d_in[1];
  const int*   dst    = (const int*)  d_in[2];
  const float* W      = (const float*)d_in[3];
  const float* attn_l = (const float*)d_in[4];
  const float* attn_r = (const float*)d_in[5];
  float* out = (float*)d_out;

  // workspace layout (~83 MB)
  float*        ft    = (float*)d_ws;                                  // N*128
  float*        el    = ft + (size_t)N_NODES * HD;                     // N*4
  float*        er    = el + (size_t)N_NODES * NUM_HEADS;              // N*4
  unsigned int* emaxk = (unsigned int*)(er + (size_t)N_NODES * NUM_HEADS); // N*4
  float*        esum  = (float*)(emaxk + (size_t)N_NODES * NUM_HEADS); // N*4
  float*        esc   = esum + (size_t)N_NODES * NUM_HEADS;            // E*4 (scores, then exp in-place)

  gat_init<<<(N_NODES * NUM_HEADS + 255) / 256, 256, 0, stream>>>(emaxk, esum);
  gat_gemm_wmma<<<(N_NODES + 63) / 64, 128, 0, stream>>>(feat, W, ft);
  gat_elr<<<(N_NODES * NUM_HEADS + 255) / 256, 256, 0, stream>>>(ft, attn_l, attn_r, el, er);
  gat_edge_score<<<(N_EDGES + 255) / 256, 256, 0, stream>>>(src, dst, el, er, esc, emaxk);
  gat_edge_exp<<<(N_EDGES + 255) / 256, 256, 0, stream>>>(dst, emaxk, esc, esum);
  gat_aggregate<<<(N_NODES * 32 + 255) / 256, 256, 0, stream>>>(src, dst, ft, esc, esum, out);
}
